// DiffPool_57071525429590
// MI455X (gfx1250) — matrix operand order repaired and checked by reading, not compiled, not run on hardware
//
#include <hip/hip_runtime.h>
#include <hip/hip_bf16.h>
#include <math.h>

typedef __attribute__((ext_vector_type(16))) _Float16 v16h;
typedef __attribute__((ext_vector_type(8)))  _Float16 v8h;
typedef __attribute__((ext_vector_type(8)))  float    v8f;

#define HDIM 128
#define NB   64
#define NPG  1024                 // nodes per graph
#define NTOT (NB * NPG)           // 65536
#define K1   820
#define K2   656
#define N1   (NB * K1)            // 52480
#define N2   (NB * K2)            // 41984
#define WPAD 136                  // LDS row stride in halves (272B: 16B-aligned, conflict-softened)

// ---------------------------------------------------------------------------
// WMMA fragment index helpers (CDNA5 wave32, 16-bit 16x32 A / 32x16 B)
// A: lane half selects contiguous K block 0-15 / 16-31; element h -> K=kbase+h
// B: lane half selects contiguous K block 0-15 / 16-31; element h -> K=kblk+h
// ---------------------------------------------------------------------------
__device__ __forceinline__ int wmma_ka(int lane, int h) {
    int kbase = (lane & 16) ? 16 : 0;
    return kbase + h;
}

// Gather one 16x16 B fragment from LDS-transposed f16 weights: two ds_load_b128
__device__ __forceinline__ v16h lds_bfrag(const _Float16* lw, int c, int kk, int kblk) {
    const v8h* bp = (const v8h*)(lw + c * WPAD + kk + kblk);
    v8h b0 = bp[0];
    v8h b1 = bp[1];
    v16h b;
#pragma unroll
    for (int h = 0; h < 8; ++h) { b[h] = b0[h]; b[h + 8] = b1[h]; }
    return b;
}

// ---------------------------------------------------------------------------
// Edge scatter-mean accumulation: one wave32 per edge.
// Lane owns 4 consecutive feats (float4 coalesced load, 4 atomic f32 adds).
// ---------------------------------------------------------------------------
__global__ void scatter_sum_kernel(const float* __restrict__ x,
                                   const int* __restrict__ src,
                                   const int* __restrict__ dst,
                                   float* __restrict__ sum,
                                   float* __restrict__ cnt, int nedges) {
    int e = blockIdx.x * (blockDim.x >> 5) + (threadIdx.x >> 5);
    if (e >= nedges) return;
    int lane = threadIdx.x & 31;
    int s = src[e], d = dst[e];
    const float4* xs = (const float4*)(x + (size_t)s * HDIM) + lane;
    float* sd = sum + (size_t)d * HDIM + lane * 4;
    float4 v = *xs;
    atomicAdd(&sd[0], v.x);
    atomicAdd(&sd[1], v.y);
    atomicAdd(&sd[2], v.z);
    atomicAdd(&sd[3], v.w);
    if (lane == 0) atomicAdd(&cnt[d], 1.0f);
}

// Same, but edges remapped through `inv` (pooled graph); dropped edges skip.
__global__ void scatter_sum_pooled_kernel(const float* __restrict__ x,
                                          const int* __restrict__ src,
                                          const int* __restrict__ dst,
                                          const int* __restrict__ inv,
                                          float* __restrict__ sum,
                                          float* __restrict__ cnt, int nedges) {
    int e = blockIdx.x * (blockDim.x >> 5) + (threadIdx.x >> 5);
    if (e >= nedges) return;
    int lane = threadIdx.x & 31;
    int ns = inv[src[e]], nd = inv[dst[e]];
    if (ns < 0 || nd < 0) return;
    const float4* xs = (const float4*)(x + (size_t)ns * HDIM) + lane;
    float* sd = sum + (size_t)nd * HDIM + lane * 4;
    float4 v = *xs;
    atomicAdd(&sd[0], v.x);
    atomicAdd(&sd[1], v.y);
    atomicAdd(&sd[2], v.z);
    atomicAdd(&sd[3], v.w);
    if (lane == 0) atomicAdd(&cnt[nd], 1.0f);
}

// ---------------------------------------------------------------------------
// Fused SAGE update (WMMA): out = relu( (sum/max(cnt,1)) @ Wl + bl + x @ Wr )
// Block = 32x8 (8 waves); wave w -> 16x16 tile (row tile blockIdx.x, col tile w)
// Weights staged per-block into LDS (f16, transposed, padded) in two phases.
// ---------------------------------------------------------------------------
__global__ void sage_update_kernel(const float* __restrict__ sum,
                                   const float* __restrict__ cnt,
                                   const float* __restrict__ x,
                                   const float* __restrict__ Wl,
                                   const float* __restrict__ bl,
                                   const float* __restrict__ Wr,
                                   float* __restrict__ out) {
    __shared__ _Float16 lw[HDIM * WPAD];   // 34.8 KB
    int lane = threadIdx.x;
    int t = threadIdx.y * 32 + lane;
    int row0 = blockIdx.x * 16;
    int col0 = threadIdx.y * 16;
    int r = row0 + (lane & 15);
    int c = col0 + (lane & 15);
    int kblk = (lane & 16) ? 16 : 0;
    float icnt = 1.0f / fmaxf(cnt[r], 1.0f);
    const float* srow = sum + (size_t)r * HDIM;
    const float* xrow = x + (size_t)r * HDIM;

    v8f acc = {};

    // ---- phase 1: stage Wl^T, accumulate mean @ Wl ----
    for (int i = t; i < HDIM * HDIM; i += 256) {
        int k = i >> 7, cc = i & 127;
        lw[cc * WPAD + k] = (_Float16)Wl[i];
    }
    __syncthreads();
    for (int kk = 0; kk < HDIM; kk += 32) {
        v16h a;
#pragma unroll
        for (int h = 0; h < 16; ++h)
            a[h] = (_Float16)(srow[kk + wmma_ka(lane, h)] * icnt);
        v16h b = lds_bfrag(lw, c, kk, kblk);
        acc = __builtin_amdgcn_wmma_f32_16x16x32_f16(false, a, false, b,
                                                     (short)0, acc, false, false);
    }
    __syncthreads();

    // ---- phase 2: stage Wr^T, accumulate x @ Wr ----
    for (int i = t; i < HDIM * HDIM; i += 256) {
        int k = i >> 7, cc = i & 127;
        lw[cc * WPAD + k] = (_Float16)Wr[i];
    }
    __syncthreads();
    for (int kk = 0; kk < HDIM; kk += 32) {
        v16h a;
#pragma unroll
        for (int h = 0; h < 16; ++h)
            a[h] = (_Float16)xrow[kk + wmma_ka(lane, h)];
        v16h b = lds_bfrag(lw, c, kk, kblk);
        acc = __builtin_amdgcn_wmma_f32_16x16x32_f16(false, a, false, b,
                                                     (short)0, acc, false, false);
    }

    float bias = bl[c];
#pragma unroll
    for (int v = 0; v < 8; ++v) {
        int m = row0 + v + ((lane & 16) ? 8 : 0);
        float val = acc[v] + bias;
        out[(size_t)m * HDIM + c] = val > 0.0f ? val : 0.0f;
    }
}

// ---------------------------------------------------------------------------
// score = tanh( x . w / ||w|| ), one 128-thread block per node
// ---------------------------------------------------------------------------
__global__ void score_kernel(const float* __restrict__ x,
                             const float* __restrict__ w,
                             float* __restrict__ score) {
    __shared__ float sdot[HDIM];
    __shared__ float snrm[HDIM];
    int node = blockIdx.x;
    int f = threadIdx.x;
    float wv = w[f];
    sdot[f] = x[(size_t)node * HDIM + f] * wv;
    snrm[f] = wv * wv;
    __syncthreads();
    for (int s = HDIM / 2; s > 0; s >>= 1) {
        if (f < s) { sdot[f] += sdot[f + s]; snrm[f] += snrm[f + s]; }
        __syncthreads();
    }
    if (f == 0) score[node] = tanhf(sdot[0] / sqrtf(snrm[0]));
}

// ---------------------------------------------------------------------------
// TopK per graph (block of 1024 threads = one graph). Exact rank selection
// (tie-break by smaller index, matching lax.top_k), compact via LDS scan.
// ---------------------------------------------------------------------------
__global__ void topk_kernel(const float* __restrict__ score, int n_per, int k,
                            int* __restrict__ perm, int* __restrict__ inv) {
    __shared__ float s[1024];
    __shared__ int pfx[1024];
    __shared__ int keep[1024];
    int b = blockIdx.x;
    int i = threadIdx.x;
    s[i] = (i < n_per) ? score[b * n_per + i] : -1.0e30f;
    __syncthreads();
    int kp = 0;
    if (i < n_per) {
        float si = s[i];
        int rank = 0;
        for (int j = 0; j < n_per; ++j) {
            float sj = s[j];
            rank += (sj > si) || (sj == si && j < i);
        }
        kp = (rank < k) ? 1 : 0;
    }
    keep[i] = kp;
    pfx[i] = kp;
    __syncthreads();
    for (int off = 1; off < 1024; off <<= 1) {
        int v = pfx[i];
        int add = (i >= off) ? pfx[i - off] : 0;
        __syncthreads();
        pfx[i] = v + add;
        __syncthreads();
    }
    if (i < n_per) {
        if (keep[i]) {
            int p = pfx[i] - 1;           // inclusive scan -> position
            perm[b * k + p] = b * n_per + i;
            inv[b * n_per + i] = b * k + p;
        } else {
            inv[b * n_per + i] = -1;
        }
    }
}

// gated gather: out[new] = x[perm[new]] * score[perm[new]]
__global__ void gate_kernel(const float* __restrict__ x,
                            const float* __restrict__ score,
                            const int* __restrict__ perm,
                            float* __restrict__ out) {
    int id = blockIdx.x;
    int f = threadIdx.x;
    int old = perm[id];
    out[(size_t)id * HDIM + f] = x[(size_t)old * HDIM + f] * score[old];
}

// readout: [B, 2H] = concat(mean, max) over k kept nodes
__global__ void readout_kernel(const float* __restrict__ h, int k,
                               float* __restrict__ out) {
    int b = blockIdx.x;
    int f = threadIdx.x;
    float sm = 0.0f, mx = -3.0e38f;
    const float* base = h + (size_t)b * k * HDIM + f;
    for (int i = 0; i < k; ++i) {
        if (i + 16 < k)
            __builtin_prefetch(base + (size_t)(i + 16) * HDIM, 0, 1);
        float v = base[(size_t)i * HDIM];
        sm += v;
        mx = fmaxf(mx, v);
    }
    out[b * (2 * HDIM) + f] = sm / (float)k;
    out[b * (2 * HDIM) + HDIM + f] = mx;
}

// ---------------------------------------------------------------------------
// MLP head layer 1 (WMMA): z = relu((x1+x2) @ fW1 + fb1); rows=64, K=256, N=128
// fW1 staged to LDS in two K-phases of 128.
// ---------------------------------------------------------------------------
__global__ void mlp1_kernel(const float* __restrict__ x1,
                            const float* __restrict__ x2,
                            const float* __restrict__ W,
                            const float* __restrict__ bias,
                            float* __restrict__ z) {
    __shared__ _Float16 lw[HDIM * WPAD];
    int lane = threadIdx.x;
    int t = threadIdx.y * 32 + lane;
    int row0 = blockIdx.x * 16;
    int col0 = threadIdx.y * 16;
    int r = row0 + (lane & 15);
    int c = col0 + (lane & 15);
    int kblk = (lane & 16) ? 16 : 0;
    v8f acc = {};

    for (int kphase = 0; kphase < 2; ++kphase) {
        int kbase = kphase * HDIM;
        for (int i = t; i < HDIM * HDIM; i += 256) {
            int k = i >> 7, cc = i & 127;
            lw[cc * WPAD + k] = (_Float16)W[(size_t)(kbase + k) * HDIM + cc];
        }
        __syncthreads();
        for (int kk = 0; kk < HDIM; kk += 32) {
            v16h a;
#pragma unroll
            for (int h = 0; h < 16; ++h) {
                int ka = kbase + kk + wmma_ka(lane, h);
                a[h] = (_Float16)(x1[r * 2 * HDIM + ka] + x2[r * 2 * HDIM + ka]);
            }
            v16h b = lds_bfrag(lw, c, kk, kblk);
            acc = __builtin_amdgcn_wmma_f32_16x16x32_f16(false, a, false, b,
                                                         (short)0, acc, false, false);
        }
        __syncthreads();
    }
    float bv = bias[c];
#pragma unroll
    for (int v = 0; v < 8; ++v) {
        int m = row0 + v + ((lane & 16) ? 8 : 0);
        float val = acc[v] + bv;
        z[m * HDIM + c] = val > 0.0f ? val : 0.0f;
    }
}

// MLP head layer 2: out[64,10] = z @ fW2 + fb2 (tiny; scalar)
__global__ void mlp2_kernel(const float* __restrict__ z,
                            const float* __restrict__ W,
                            const float* __restrict__ bias,
                            float* __restrict__ out, int ncls) {
    int idx = blockIdx.x * blockDim.x + threadIdx.x;
    if (idx >= NB * ncls) return;
    int b = idx / ncls, c = idx % ncls;
    float acc = bias[c];
    for (int k = 0; k < HDIM; ++k) acc += z[b * HDIM + k] * W[k * ncls + c];
    out[idx] = acc;
}

// ---------------------------------------------------------------------------
extern "C" void kernel_launch(void* const* d_in, const int* in_sizes, int n_in,
                              void* d_out, int out_size, void* d_ws, size_t ws_size,
                              hipStream_t stream) {
    const float* x    = (const float*)d_in[0];
    const int*   ei   = (const int*)d_in[1];
    const float* Wl1  = (const float*)d_in[3];
    const float* bl1  = (const float*)d_in[4];
    const float* Wr1  = (const float*)d_in[5];
    const float* Wl2  = (const float*)d_in[6];
    const float* bl2  = (const float*)d_in[7];
    const float* Wr2  = (const float*)d_in[8];
    const float* pw1  = (const float*)d_in[9];
    const float* pw2  = (const float*)d_in[10];
    const float* fW1  = (const float*)d_in[11];
    const float* fb1  = (const float*)d_in[12];
    const float* fW2  = (const float*)d_in[13];
    const float* fb2  = (const float*)d_in[14];

    const int E = in_sizes[1] / 2;
    const int* src = ei;
    const int* dst = ei + E;

    // workspace carve (bump allocator, 256B aligned)
    char* p = (char*)d_ws;
    auto carve = [&](size_t bytes) -> void* {
        void* r = (void*)p;
        p += (bytes + 255) & ~(size_t)255;
        return r;
    };
    float* sum1   = (float*)carve((size_t)NTOT * HDIM * 4);
    float* cnt1   = (float*)carve((size_t)NTOT * 4);
    float* h1     = (float*)carve((size_t)NTOT * HDIM * 4);
    float* score1 = (float*)carve((size_t)NTOT * 4);
    int*   perm1  = (int*)carve((size_t)N1 * 4);
    int*   inv1   = (int*)carve((size_t)NTOT * 4);
    float* h1p    = (float*)carve((size_t)N1 * HDIM * 4);
    float* x1r    = (float*)carve((size_t)NB * 2 * HDIM * 4);
    float* sum2   = (float*)carve((size_t)N1 * HDIM * 4);
    float* cnt2   = (float*)carve((size_t)N1 * 4);
    float* h2     = (float*)carve((size_t)N1 * HDIM * 4);
    float* score2 = (float*)carve((size_t)N1 * 4);
    int*   perm2  = (int*)carve((size_t)N2 * 4);
    int*   inv2   = (int*)carve((size_t)N1 * 4);
    float* h2p    = (float*)carve((size_t)N2 * HDIM * 4);
    float* x2r    = (float*)carve((size_t)NB * 2 * HDIM * 4);
    float* zbuf   = (float*)carve((size_t)NB * HDIM * 4);

    // zero accumulators (every call: deterministic)
    hipMemsetAsync(sum1, 0, (size_t)NTOT * HDIM * 4, stream);
    hipMemsetAsync(cnt1, 0, (size_t)NTOT * 4, stream);
    hipMemsetAsync(sum2, 0, (size_t)N1 * HDIM * 4, stream);
    hipMemsetAsync(cnt2, 0, (size_t)N1 * 4, stream);

    dim3 wblk(32, 8);  // 8 waves per block for WMMA kernels
    const int epb = 256 / 32;  // edges per block (wave per edge)

    // ---- SAGE layer 1 ----
    scatter_sum_kernel<<<(E + epb - 1) / epb, 256, 0, stream>>>(
        x, src, dst, sum1, cnt1, E);
    sage_update_kernel<<<NTOT / 16, wblk, 0, stream>>>(
        sum1, cnt1, x, Wl1, bl1, Wr1, h1);

    // ---- TopK pool 1 + readout ----
    score_kernel<<<NTOT, HDIM, 0, stream>>>(h1, pw1, score1);
    topk_kernel<<<NB, 1024, 0, stream>>>(score1, NPG, K1, perm1, inv1);
    gate_kernel<<<N1, HDIM, 0, stream>>>(h1, score1, perm1, h1p);
    readout_kernel<<<NB, HDIM, 0, stream>>>(h1p, K1, x1r);

    // ---- SAGE layer 2 (edges remapped through inv1) ----
    scatter_sum_pooled_kernel<<<(E + epb - 1) / epb, 256, 0, stream>>>(
        h1p, src, dst, inv1, sum2, cnt2, E);
    sage_update_kernel<<<N1 / 16, wblk, 0, stream>>>(
        sum2, cnt2, h1p, Wl2, bl2, Wr2, h2);

    // ---- TopK pool 2 + readout ----
    score_kernel<<<N1, HDIM, 0, stream>>>(h2, pw2, score2);
    topk_kernel<<<NB, 1024, 0, stream>>>(score2, K1, K2, perm2, inv2);
    gate_kernel<<<N2, HDIM, 0, stream>>>(h2, score2, perm2, h2p);
    readout_kernel<<<NB, HDIM, 0, stream>>>(h2p, K2, x2r);

    // ---- MLP head ----
    mlp1_kernel<<<NB / 16, wblk, 0, stream>>>(x1r, x2r, fW1, fb1, zbuf);
    mlp2_kernel<<<(NB * 10 + 255) / 256, 256, 0, stream>>>(zbuf, fW2, fb2,
                                                           (float*)d_out, 10);
}
